// BatchLoreAttentionLayer_25151328485704
// MI455X (gfx1250) — matrix operand order repaired and checked by reading, not compiled, not run on hardware
//
#include <hip/hip_runtime.h>
#include <hip/hip_bf16.h>
#include <math.h>

// Problem constants (from reference): B=2048, L=128, D=256
#define BATCH 2048
#define LSEQ  128
#define DDIM  256
#define XT_STRIDE 130   // padded f16 stride for transposed X (bank-spread)

typedef __attribute__((ext_vector_type(16))) _Float16 v16h;
typedef __attribute__((ext_vector_type(4)))  _Float16 v4h;
typedef __attribute__((ext_vector_type(8)))  float    v8f;

// ---------------------------------------------------------------------------
// Fragment loaders (wave32, CDNA5 WMMA 16x16x32 f16 layouts, ISA 7.12.2)
// A (16x32 f16): lanes 0-15 / 16-31 both hold M=0..15.
//   VGPR v in 0..7 : K = 16*(lane>=16) + 2v, 2v+1 (contiguous order used here)
// B (32x16 f16, this code's k order, matching A): lane = n + 16*g ; VGPR v holds K = 16*g + 2v, 2v+1 ; col n.
// C/D (16x16 f32): lane = n + 16*(m>=8) ; VGPR r holds row m = r + 8*(lane>=16).
// ---------------------------------------------------------------------------

__device__ __forceinline__ v16h load_a_frag(const _Float16* lds, int row0,
                                            int stride, int k0) {
  const int lane = threadIdx.x & 31;
  const int m = lane & 15;
  const int g = lane >> 4;
  const _Float16* r = lds + (row0 + m) * stride + k0 + 16 * g;
  v16h a;
#pragma unroll
  for (int v = 0; v < 8; ++v) {
    const int k = 2 * v;
    a[2 * v]     = r[k];
    a[2 * v + 1] = r[k + 1];
  }
  return a;
}

// B = W^T chunk for projections from pre-converted f16 weights (row-major
// [e][d]). Lane n reads 16 contiguous halves of W row (e0+n) -> wide b128
// global loads, no conversion VALU in the hot loop.
__device__ __forceinline__ v16h load_w_frag(const _Float16* __restrict__ Wh,
                                            int e0, int d0) {
  const int lane = threadIdx.x & 31;
  const int n = lane & 15;
  const int g = lane >> 4;
  const _Float16* r = Wh + (size_t)(e0 + n) * DDIM + d0 + 16 * g;
  v16h b;
#pragma unroll
  for (int i = 0; i < 16; ++i) b[i] = r[i];
  return b;
}

// Generic "B from a transposed-layout LDS panel": source is stored so that the
// B-matrix column index n selects a row (row0+n) and the WMMA K index runs
// contiguously along that row. Lane n reads 16 contiguous halves at
// [row0+n][k0 + 16g] -> two ds_load_b128 per fragment.
// Used for: scores B = K^T (from row-major Kh, K index = embedding dim) and
// attended B = X (from transposed XhT, K index = sequence dim).
__device__ __forceinline__ v16h load_bT_frag(const _Float16* lds, int row0,
                                             int stride, int k0) {
  const int lane = threadIdx.x & 31;
  const int n = lane & 15;
  const int g = lane >> 4;
  const _Float16* r = lds + (row0 + n) * stride + k0 + 16 * g;
  v16h b;
#pragma unroll
  for (int i = 0; i < 16; ++i) b[i] = r[i];
  return b;
}

__device__ __forceinline__ v8f wmma_f16(v16h a, v16h b, v8f c) {
  return __builtin_amdgcn_wmma_f32_16x16x32_f16(false, a, false, b,
                                                (short)0, c, false, false);
}

// ---------------------------------------------------------------------------
// Prepass: convert q_w / k_w (f32, 256x256 each) to f16 once into workspace.
// ---------------------------------------------------------------------------
extern "C" __global__ __launch_bounds__(256)
void convert_weights(const float* __restrict__ qw,
                     const float* __restrict__ kw,
                     _Float16* __restrict__ wq_h,
                     _Float16* __restrict__ wk_h) {
  const int i = blockIdx.x * 256 + threadIdx.x;   // 0 .. 16383
  float4 a = ((const float4*)qw)[i];
  float4 b = ((const float4*)kw)[i];
  v4h ha = { (_Float16)a.x, (_Float16)a.y, (_Float16)a.z, (_Float16)a.w };
  v4h hb = { (_Float16)b.x, (_Float16)b.y, (_Float16)b.z, (_Float16)b.w };
  ((v4h*)wq_h)[i] = ha;
  ((v4h*)wk_h)[i] = hb;
}

// ---------------------------------------------------------------------------
// One workgroup per batch element. 256 threads = 8 wave32; wave w owns rows
// [16w, 16w+16).
// Dynamic LDS carve (total 296,448 B <= 320 KB WGP limit):
//   [0      , 65536 )  Xh   f16 [128][256]           (row-major X)
//   [65536  , 131072)  Qh   f16 [128][256]   } overlaid later by
//   [131072 , 196608)  Kh   f16 [128][256]   } attf f32 [128][256]
//   [196608 , 229376)  Ph   f16 [128][128]
//   [229376 , 295936)  XhT  f16 [256][130]           (transposed X, padded)
//   [295936 , 296448)  validf f32 [128]
// ---------------------------------------------------------------------------
extern "C" __global__ __launch_bounds__(256)
void batch_lore_attention(const float* __restrict__ emb,
                          const unsigned char* __restrict__ mask,
                          const _Float16* __restrict__ wq_h,
                          const float* __restrict__ qb,
                          const _Float16* __restrict__ wk_h,
                          const float* __restrict__ kb,
                          float* __restrict__ out) {
  extern __shared__ char smem[];
  _Float16* Xh  = (_Float16*)(smem);
  _Float16* Qh  = (_Float16*)(smem + 65536);
  _Float16* Kh  = (_Float16*)(smem + 131072);
  _Float16* Ph  = (_Float16*)(smem + 196608);
  _Float16* XhT = (_Float16*)(smem + 229376);
  float*  validf = (float*)(smem + 295936);
  float*  attf   = (float*)(smem + 65536);   // overlays Qh+Kh (dead by then)

  const int b    = blockIdx.x;
  const int tid  = threadIdx.x;
  const int lane = tid & 31;
  const int w    = tid >> 5;
  const int n    = lane & 15;
  const int g    = lane >> 4;
  const int row0 = 16 * w;

  // ---- Stage 0: stage X (f32->f16) row-major + transposed into LDS ----
  {
    const float4* X4 = (const float4*)(emb + (size_t)b * LSEQ * DDIM);
#pragma unroll 4
    for (int i = tid; i < (LSEQ * DDIM) / 4; i += 256) {
      float4 v = X4[i];
      v4h h = { (_Float16)v.x, (_Float16)v.y, (_Float16)v.z, (_Float16)v.w };
      ((v4h*)Xh)[i] = h;
      const int l = i >> 6;            // row    (4i / 256)
      const int d = (i & 63) << 2;     // column (4i % 256)
      XhT[(size_t)(d + 0) * XT_STRIDE + l] = h[0];
      XhT[(size_t)(d + 1) * XT_STRIDE + l] = h[1];
      XhT[(size_t)(d + 2) * XT_STRIDE + l] = h[2];
      XhT[(size_t)(d + 3) * XT_STRIDE + l] = h[3];
    }
    if (tid < LSEQ)
      validf[tid] = mask[(size_t)b * LSEQ + tid] ? 0.0f : 1.0f;
  }
  __syncthreads();

  // ---- Stage 1: Q = Xh @ Wq^T + bq ; K = Xh @ Wk^T + bk (f16 out in LDS) ----
  {
    v16h afr[8];
#pragma unroll
    for (int kk = 0; kk < 8; ++kk)
      afr[kk] = load_a_frag(Xh, row0, DDIM, 32 * kk);

    for (int t = 0; t < 16; ++t) {
      v8f aq = {};
      v8f ak = {};
#pragma unroll
      for (int kk = 0; kk < 8; ++kk) {
        aq = wmma_f16(afr[kk], load_w_frag(wq_h, 16 * t, 32 * kk), aq);
        ak = wmma_f16(afr[kk], load_w_frag(wk_h, 16 * t, 32 * kk), ak);
      }
      const int col = 16 * t + n;
      const float bq = qb[col];
      const float bk = kb[col];
#pragma unroll
      for (int r = 0; r < 8; ++r) {
        const int row = row0 + r + 8 * g;
        Qh[(size_t)row * DDIM + col] = (_Float16)(aq[r] + bq);
        Kh[(size_t)row * DDIM + col] = (_Float16)(ak[r] + bk);
      }
    }
  }
  __syncthreads();

  // ---- Stage 2: S = Q K^T / sqrt(D), mask, softmax -> Ph (f16) ----
  {
    v16h aq[4];
#pragma unroll
    for (int kk = 0; kk < 4; ++kk)
      aq[kk] = load_a_frag(Qh, row0, DDIM, 32 * kk);

    v8f s[8];
#pragma unroll
    for (int t = 0; t < 8; ++t) s[t] = v8f{};
#pragma unroll
    for (int t = 0; t < 8; ++t)
#pragma unroll
      for (int kk = 0; kk < 4; ++kk)
        s[t] = wmma_f16(aq[kk], load_bT_frag(Kh, 16 * t, DDIM, 32 * kk), s[t]);

    // scale + key-padding mask (key index = 16t + n for this lane)
    const float kscale = 0.0625f;  // 1/sqrt(256)
#pragma unroll
    for (int t = 0; t < 8; ++t) {
      const float vkey = validf[16 * t + n];
#pragma unroll
      for (int r = 0; r < 8; ++r) {
        float val = s[t][r] * kscale;
        s[t][r] = (vkey > 0.0f) ? val : -INFINITY;
      }
    }
    // row-wise softmax: rows live across lanes {0..15} (g=0) / {16..31} (g=1)
#pragma unroll
    for (int r = 0; r < 8; ++r) {
      float m = -INFINITY;
#pragma unroll
      for (int t = 0; t < 8; ++t) m = fmaxf(m, s[t][r]);
      m = fmaxf(m, __shfl_xor(m, 1, 32));
      m = fmaxf(m, __shfl_xor(m, 2, 32));
      m = fmaxf(m, __shfl_xor(m, 4, 32));
      m = fmaxf(m, __shfl_xor(m, 8, 32));
      float sum = 0.0f;
#pragma unroll
      for (int t = 0; t < 8; ++t) {
        float e = __expf(s[t][r] - m);
        s[t][r] = e;
        sum += e;
      }
      sum += __shfl_xor(sum, 1, 32);
      sum += __shfl_xor(sum, 2, 32);
      sum += __shfl_xor(sum, 4, 32);
      sum += __shfl_xor(sum, 8, 32);
      const float inv = 1.0f / sum;
      const int row = row0 + r + 8 * g;
#pragma unroll
      for (int t = 0; t < 8; ++t)
        Ph[(size_t)row * LSEQ + 16 * t + n] = (_Float16)(s[t][r] * inv);
    }
  }
  __syncthreads();

  // ---- Stage 3: attended = P @ X  (B-frags from transposed XhT) ----
  {
    v16h ap[4];
#pragma unroll
    for (int kk = 0; kk < 4; ++kk)
      ap[kk] = load_a_frag(Ph, row0, LSEQ, 32 * kk);

    for (int t = 0; t < 16; ++t) {
      v8f ac = {};
#pragma unroll
      for (int kk = 0; kk < 4; ++kk)
        ac = wmma_f16(ap[kk], load_bT_frag(XhT, 16 * t, XT_STRIDE, 32 * kk), ac);
      const int col = 16 * t + n;
#pragma unroll
      for (int r = 0; r < 8; ++r)
        attf[(size_t)(row0 + r + 8 * g) * DDIM + col] = ac[r];
    }
  }
  __syncthreads();

  // ---- Stage 4: masked mean-pool over valid queries + tanh ----
  {
    float acc = 0.0f, cnt = 0.0f;
#pragma unroll 4
    for (int l = 0; l < LSEQ; ++l) {
      const float v = validf[l];
      acc += attf[(size_t)l * DDIM + tid] * v;
      cnt += v;
    }
    out[(size_t)b * DDIM + tid] = tanhf(acc / fmaxf(cnt, 1.0f));
  }
}

extern "C" void kernel_launch(void* const* d_in, const int* in_sizes, int n_in,
                              void* d_out, int out_size, void* d_ws, size_t ws_size,
                              hipStream_t stream) {
  (void)in_sizes; (void)n_in; (void)out_size; (void)ws_size;
  const float*         emb  = (const float*)d_in[0];
  const unsigned char* mask = (const unsigned char*)d_in[1];  // bool mask
  const float*         qw   = (const float*)d_in[2];
  const float*         qb   = (const float*)d_in[3];
  const float*         kw   = (const float*)d_in[4];
  const float*         kb   = (const float*)d_in[5];
  float*               out  = (float*)d_out;

  // Workspace: f16 copies of the two weight matrices (128 KB each).
  _Float16* wq_h = (_Float16*)d_ws;
  _Float16* wk_h = (_Float16*)((char*)d_ws + (size_t)DDIM * DDIM * sizeof(_Float16));

  convert_weights<<<dim3((DDIM * DDIM / 4) / 256), dim3(256), 0, stream>>>(
      qw, kw, wq_h, wk_h);

  const size_t shmem = 296448;  // see LDS carve above
  batch_lore_attention<<<dim3(BATCH), dim3(256), shmem, stream>>>(
      emb, mask, wq_h, qb, wk_h, kb, out);
}
